// Pooler_77335181131834
// MI455X (gfx1250) — compile-verified, matrix operand order checked by reading, not run on hardware
//
#include <hip/hip_runtime.h>
#include <hip/hip_bf16.h>

typedef __attribute__((ext_vector_type(16))) __bf16 v16bf;
typedef __attribute__((ext_vector_type(8)))  float  v8f;
typedef int v4i __attribute__((vector_size(16)));   // matches builtin param type

#define B_   32
#define L_   2048
#define D_   1024          // DQ = DK = NH*DH
#define NH_  16
#define DH_  64
#define M_KV (B_ * L_)     // 65536 rows of k (b*L + l)
#define NKV  2048          // Wk rows (0..1023) concat Wv rows (1024..2047)

#ifndef __has_builtin
#define __has_builtin(x) 0
#endif
#if __has_builtin(__builtin_amdgcn_global_load_async_to_lds_b128)
#define ASYNC_LDS 1
#else
#define ASYNC_LDS 0
#endif

#define AS1 __attribute__((address_space(1)))
#define AS3 __attribute__((address_space(3)))

#if __has_builtin(__builtin_amdgcn_s_wait_asynccnt)
#define WAIT_ASYNC(n) __builtin_amdgcn_s_wait_asynccnt(n)
#else
#define WAIT_ASYNC(n) asm volatile("s_wait_asynccnt %0" ::"n"(n) : "memory")
#endif

union FragBF { uint4 q[2]; v16bf v; };

__device__ __forceinline__ unsigned short f32_to_bf16(float f) {
  unsigned int u = __float_as_uint(f);
  u += 0x7FFFu + ((u >> 16) & 1u);     // round-to-nearest-even
  return (unsigned short)(u >> 16);
}
__device__ __forceinline__ float b2f_lo(unsigned int w) { return __uint_as_float(w << 16); }
__device__ __forceinline__ float b2f_hi(unsigned int w) { return __uint_as_float(w & 0xffff0000u); }

// ---------------------------------------------------------------- convert
__global__ void cvt_f32_bf16(const float* __restrict__ src,
                             unsigned short* __restrict__ dst, int n4) {
  int i = blockIdx.x * blockDim.x + threadIdx.x;
  int stride = gridDim.x * blockDim.x;
  const float4* s4 = (const float4*)src;
  ushort4* d4 = (ushort4*)dst;
  for (; i < n4; i += stride) {
    float4 f = s4[i];
    ushort4 o;
    o.x = f32_to_bf16(f.x); o.y = f32_to_bf16(f.y);
    o.z = f32_to_bf16(f.z); o.w = f32_to_bf16(f.w);
    d4[i] = o;
  }
}

// ---------------------------------------------------------------- Q projection
// qs[32,1024] = q_bf16[32,1024] @ Wq_bf16^T + bq, one wave per 16x64 tile
__global__ void __launch_bounds__(32)
qproj_wmma(const unsigned short* __restrict__ qbf,
           const unsigned short* __restrict__ wqbf,
           const float* __restrict__ bq,
           float* __restrict__ qs) {
  const int lane = threadIdx.x;
  const int tM   = blockIdx.x & 1;          // rows 0..15 / 16..31
  const int n0   = (blockIdx.x >> 1) * 64;  // 16 N-tiles of 64
  const int lrow = lane & 15;
  const int kb   = (lane >> 4) * 8;         // K sub-base (0 or 8)

  v8f acc[4];
#pragma unroll
  for (int j = 0; j < 4; ++j) acc[j] = (v8f){};

  const unsigned short* arow = qbf + (tM * 16 + lrow) * D_;
  for (int k0 = 0; k0 < D_; k0 += 32) {
    FragBF a;
    a.q[0] = *(const uint4*)(arow + k0 + kb);        // K = kb .. kb+7
    a.q[1] = *(const uint4*)(arow + k0 + kb + 16);   // K = kb+16 .. kb+23
#pragma unroll
    for (int j = 0; j < 4; ++j) {
      const unsigned short* brow = wqbf + (size_t)(n0 + j * 16 + lrow) * D_;
      FragBF b;
      b.q[0] = *(const uint4*)(brow + k0 + kb);
      b.q[1] = *(const uint4*)(brow + k0 + kb + 16);
      acc[j] = __builtin_amdgcn_wmma_f32_16x16x32_bf16(
          false, a.v, false, b.v, (short)0, acc[j], false, false);
    }
  }
#pragma unroll
  for (int j = 0; j < 4; ++j) {
    int n = n0 + j * 16 + lrow;
    float bias = bq[n];
#pragma unroll
    for (int r = 0; r < 8; ++r) {
      int m = tM * 16 + r + 8 * (lane >> 4);
      qs[m * D_ + n] = acc[j][r] + bias;
    }
  }
}

// ---------------------------------------------------------------- K/V projection
// kv[65536,2048] = k_bf16[65536,1024] @ [Wk;Wv]^T + [bk;bv], output bf16
// Double-buffered LDS pipeline; async memory->LDS when available.
#define KC 32
#define MT 64
#define NT 256
#define LSTR 40   // LDS row stride in halves (80B: 16B aligned, skews banks)
#define AUNITS (MT * 4)            // 256 x 16B for A tile
#define TUNITS ((MT + NT) * 4)     // 1280 x 16B total per stage

__global__ void __launch_bounds__(512)
kvproj_wmma(const unsigned short* __restrict__ kbf,
            const unsigned short* __restrict__ wkv,
            const float* __restrict__ bk,
            const float* __restrict__ bv,
            unsigned short* __restrict__ kvout) {
  __shared__ __align__(16) unsigned short As[2][MT * LSTR];
  __shared__ __align__(16) unsigned short Bs[2][NT * LSTR];
  const int tid  = threadIdx.x;
  const int lane = tid & 31;
  const int wave = tid >> 5;   // 0..15
  const int wm   = wave >> 2;  // M sub-tile 0..3
  const int wn   = wave & 3;   // N group of 64, 0..3
  const int mBase = blockIdx.y * MT;
  const int nBase = blockIdx.x * NT;
  const int lrow = lane & 15;
  const int kb   = (lane >> 4) * 8;

  v8f acc[4];
#pragma unroll
  for (int j = 0; j < 4; ++j) acc[j] = (v8f){};

  // stage K-chunk [k0, k0+32) of A (64 rows of k) and B (256 rows of Wkv)
  // into buffer `buf`. 1280 units of 16B; waves 0..7 issue 3 async ops,
  // waves 8..15 issue 2 (uniform within each wave).
  auto stage = [&](int k0, int buf) {
    for (int u = tid; u < TUNITS; u += 512) {
      const unsigned short* gsrc;
      unsigned short* ldst;
      if (u < AUNITS) {
        int row = u >> 2, seg = u & 3;
        gsrc = kbf + (size_t)(mBase + row) * D_ + k0 + seg * 8;
        ldst = &As[buf][row * LSTR + seg * 8];
      } else {
        int v = u - AUNITS;
        int row = v >> 2, seg = v & 3;
        gsrc = wkv + (size_t)(nBase + row) * D_ + k0 + seg * 8;
        ldst = &Bs[buf][row * LSTR + seg * 8];
      }
#if ASYNC_LDS
      __builtin_amdgcn_global_load_async_to_lds_b128(
          (AS1 v4i*)(void*)gsrc,
          (AS3 v4i*)(void*)ldst, 0, 0);
#else
      *(uint4*)ldst = *(const uint4*)gsrc;
#endif
    }
  };

  stage(0, 0);
  int buf = 0;
  for (int k0 = 0; k0 < D_; k0 += KC) {
    const int nk = k0 + KC;
    if (nk < D_) stage(nk, buf ^ 1);
#if ASYNC_LDS
    // Async loads complete in order: waiting down to this wave's next-stage
    // issue count guarantees the current stage has landed in LDS.
    if (nk < D_) {
      if (wave < 8) WAIT_ASYNC(3);
      else          WAIT_ASYNC(2);
    } else {
      WAIT_ASYNC(0);
    }
#endif
    __syncthreads();

    FragBF a;
    const unsigned short* ap = &As[buf][(wm * 16 + lrow) * LSTR + kb];
    a.q[0] = *(const uint4*)(ap);
    a.q[1] = *(const uint4*)(ap + 16);
#pragma unroll
    for (int j = 0; j < 4; ++j) {
      FragBF b;
      const unsigned short* bp = &Bs[buf][(wn * 64 + j * 16 + lrow) * LSTR + kb];
      b.q[0] = *(const uint4*)(bp);
      b.q[1] = *(const uint4*)(bp + 16);
      acc[j] = __builtin_amdgcn_wmma_f32_16x16x32_bf16(
          false, a.v, false, b.v, (short)0, acc[j], false, false);
    }
    __syncthreads();   // all waves done reading buf before it is restaged
    buf ^= 1;
  }

#pragma unroll
  for (int j = 0; j < 4; ++j) {
    int n = nBase + wn * 64 + j * 16 + lrow;
    float bias = (n < D_) ? bk[n] : bv[n - D_];
#pragma unroll
    for (int r = 0; r < 8; ++r) {
      int m = mBase + wm * 16 + r + 8 * (lane >> 4);
      kvout[(size_t)m * NKV + n] = f32_to_bf16(acc[j][r] + bias);
    }
  }
}

// ---------------------------------------------------------------- attention
// one block per (head n, batch b): scores -> softmax -> weighted V
__global__ void __launch_bounds__(256)
attn_kernel(const float* __restrict__ qs,
            const unsigned short* __restrict__ kv,
            float* __restrict__ out,
            float* __restrict__ attn) {
  const int n = blockIdx.x;
  const int b = blockIdx.y;
  const int tid = threadIdx.x;
  __shared__ float qsh[DH_];
  __shared__ float sc[L_];
  __shared__ float red[256];
  __shared__ float part[4][DH_];

  if (tid < DH_) qsh[tid] = qs[b * D_ + n * DH_ + tid];
  __syncthreads();

  float lmax = -3.0e38f;
  for (int l = tid; l < L_; l += 256) {
    const unsigned short* kp = kv + (size_t)(b * L_ + l) * NKV + n * DH_;
    float dot = 0.0f;
#pragma unroll
    for (int d0 = 0; d0 < DH_; d0 += 8) {
      uint4 u = *(const uint4*)(kp + d0);
      dot += qsh[d0 + 0] * b2f_lo(u.x) + qsh[d0 + 1] * b2f_hi(u.x)
           + qsh[d0 + 2] * b2f_lo(u.y) + qsh[d0 + 3] * b2f_hi(u.y)
           + qsh[d0 + 4] * b2f_lo(u.z) + qsh[d0 + 5] * b2f_hi(u.z)
           + qsh[d0 + 6] * b2f_lo(u.w) + qsh[d0 + 7] * b2f_hi(u.w);
    }
    float s = dot * 0.125f;   // 1/sqrt(64)
    sc[l] = s;
    lmax = fmaxf(lmax, s);
  }
  red[tid] = lmax; __syncthreads();
  for (int off = 128; off > 0; off >>= 1) {
    if (tid < off) red[tid] = fmaxf(red[tid], red[tid + off]);
    __syncthreads();
  }
  float gmax = red[0];
  __syncthreads();

  float lsum = 0.0f;
  for (int l = tid; l < L_; l += 256) {
    float e = __expf(sc[l] - gmax);
    sc[l] = e;
    lsum += e;
  }
  red[tid] = lsum; __syncthreads();
  for (int off = 128; off > 0; off >>= 1) {
    if (tid < off) red[tid] += red[tid + off];
    __syncthreads();
  }
  float inv = 1.0f / red[0];

  float* arow = attn + (size_t)(n * B_ + b) * L_;   // head-major [n*B+b, l]
  for (int l = tid; l < L_; l += 256) {
    float p = sc[l] * inv;
    sc[l] = p;
    arow[l] = p;
  }
  __syncthreads();

  const int d = tid & (DH_ - 1);
  const int g = tid >> 6;        // 4 l-groups
  float accv = 0.0f;
  for (int l = g; l < L_; l += 4) {
    unsigned short vh = kv[(size_t)(b * L_ + l) * NKV + D_ + n * DH_ + d];
    accv += sc[l] * __uint_as_float(((unsigned int)vh) << 16);
  }
  part[g][d] = accv; __syncthreads();
  if (tid < DH_)
    out[b * D_ + n * DH_ + tid] =
        part[0][tid] + part[1][tid] + part[2][tid] + part[3][tid];
}

// ---------------------------------------------------------------- launch
extern "C" void kernel_launch(void* const* d_in, const int* in_sizes, int n_in,
                              void* d_out, int out_size, void* d_ws, size_t ws_size,
                              hipStream_t stream) {
  (void)in_sizes; (void)n_in; (void)out_size; (void)ws_size;
  const float* q  = (const float*)d_in[0];
  const float* k  = (const float*)d_in[1];
  const float* Wq = (const float*)d_in[2];
  const float* bq = (const float*)d_in[3];
  const float* Wk = (const float*)d_in[4];
  const float* bk = (const float*)d_in[5];
  const float* Wv = (const float*)d_in[6];
  const float* bv = (const float*)d_in[7];

  float* out  = (float*)d_out;          // [32,1024]
  float* attn = out + B_ * D_;          // [16*32, 2048]

  // workspace carve (~409 MB)
  unsigned short* kbf  = (unsigned short*)d_ws;                 // 64M halves
  unsigned short* wkv  = kbf + (size_t)M_KV * D_;               // [2048,1024]
  unsigned short* wqb  = wkv + (size_t)NKV * D_;                // [1024,1024]
  unsigned short* qbf  = wqb + (size_t)D_ * D_;                 // [32,1024]
  float*          qsf  = (float*)(qbf + B_ * D_);               // [32,1024] f32
  unsigned short* kvbf = (unsigned short*)(qsf + B_ * D_);      // [65536,2048]

  auto nblk = [](int n4) { int b = (n4 + 255) / 256; return b > 16384 ? 16384 : b; };
  const int WW4 = D_ * D_ / 4;
  cvt_f32_bf16<<<nblk(M_KV * D_ / 4), 256, 0, stream>>>(k, kbf, M_KV * D_ / 4);
  cvt_f32_bf16<<<nblk(WW4), 256, 0, stream>>>(Wk, wkv, WW4);
  cvt_f32_bf16<<<nblk(WW4), 256, 0, stream>>>(Wv, wkv + (size_t)D_ * D_, WW4);
  cvt_f32_bf16<<<nblk(WW4), 256, 0, stream>>>(Wq, wqb, WW4);
  cvt_f32_bf16<<<nblk(B_ * D_ / 4), 256, 0, stream>>>(q, qbf, B_ * D_ / 4);

  qproj_wmma<<<dim3(32), dim3(32), 0, stream>>>(qbf, wqb, bq, qsf);
  kvproj_wmma<<<dim3(NKV / NT, M_KV / MT), dim3(512), 0, stream>>>(kbf, wkv, bk, bv, kvbf);
  attn_kernel<<<dim3(NH_, B_), dim3(256), 0, stream>>>(qsf, kvbf, out, attn);
}
